// Attention_38182259261827
// MI455X (gfx1250) — matrix operand order repaired and checked by reading, not compiled, not run on hardware
//
#include <hip/hip_runtime.h>
#include <cmath>

// ---------------------------------------------------------------------------
// Attention block for MI455X (gfx1250): bf16 WMMA (f32 accum) everywhere,
// async global->LDS staging (GLOBAL_LOAD_ASYNC_TO_LDS_B128 / ASYNCcnt) for the
// bf16 tiles in stages 2 and 3.
//   x:[4,512,2048] f32, w_qkv:[512,1536] f32, w_out:[512,512] f32, b_out:[512]
//   out:[4,512,2048] f32
// Workspace (bf16): Q[B,H,L,D] (q pre-scaled), K[B,H,L,D], V[B,H,D,L],
//                   O[B,L,H*D]  -> 4 x 8MB = 32MB
// ---------------------------------------------------------------------------

#define BATCH 4
#define CIN   512
#define LSEQ  2048
#define HEADS 8
#define DHEAD 64
#define HD    512      // HEADS*DHEAD
#define F3    1536     // 3*HD
#define QK_SCALE 0.125f  // DHEAD^-0.5

typedef __bf16 v16bf __attribute__((ext_vector_type(16)));
typedef __bf16 v8bf  __attribute__((ext_vector_type(8)));
typedef float  v8f   __attribute__((ext_vector_type(8)));

// Build a 16-element bf16 fragment from two 16-byte chunks (lowers to b128 loads).
__device__ __forceinline__ v16bf frag16(const __bf16* p0, const __bf16* p1) {
  v8bf lo = *(const v8bf*)p0;
  v8bf hi = *(const v8bf*)p1;
  v16bf r;
#pragma unroll
  for (int i = 0; i < 8; ++i) { r[i] = lo[i]; r[i + 8] = hi[i]; }
  return r;
}

// A-matrix 16x32 bf16 fragment from tile[m][k] (k contiguous, row stride ld elems).
// lanes 0-15: M=lane, K = 0..7 / 16..23 ; lanes 16-31: M=lane-16, K = 8..15 / 24..31
__device__ __forceinline__ v16bf load_frag_a(const __bf16* tile, int ld, int lane) {
  int m  = lane & 15;
  int kb = (lane & 16) ? 8 : 0;
  const __bf16* p = tile + m * ld + kb;
  return frag16(p, p + 16);
}

// B-matrix 32x16 bf16 fragment from tile[n][k] (k contiguous, row stride ld elems).
// lanes 0-15: N=lane, K = 0..7 / 16..23 ; lanes 16-31: N=lane-16, K = 8..15 / 24..31
__device__ __forceinline__ v16bf load_frag_b(const __bf16* tile, int ld, int lane) {
  int n  = lane & 15;
  int kb = (lane & 16) ? 8 : 0;
  const __bf16* p = tile + n * ld + kb;
  return frag16(p, p + 16);
}

__device__ __forceinline__ v8f wmma_bf16(v16bf a, v16bf b, v8f c) {
  return __builtin_amdgcn_wmma_f32_16x16x32_bf16(false, a, false, b, (short)0, c,
                                                 false, false);
}

// Async copy 16 bytes per lane from global to LDS (GLOBAL_LOAD_ASYNC_TO_LDS_B128,
// tracked by ASYNCcnt). lds_byte is the low-32-bit LDS address (flat addr[31:0]).
__device__ __forceinline__ void async_b128(unsigned lds_byte, const void* gaddr) {
  asm volatile("global_load_async_to_lds_b128 %0, %1, off"
               :: "v"(lds_byte), "v"((unsigned long long)(uintptr_t)gaddr)
               : "memory");
}
__device__ __forceinline__ void wait_async0() {
  asm volatile("s_wait_asynccnt 0" ::: "memory");
}
__device__ __forceinline__ unsigned lds_addr(const void* p) {
  return (unsigned)(uintptr_t)p;  // flat LDS aperture: addr[31:0] is the LDS offset
}

// ---------------------------------------------------------------------------
// Stage 1: qkv[b][f][l] = sum_c wqkv[c][f] * x[b][c][l]   (M=f 1536, N=l 2048, K=c 512)
// f32 inputs converted to bf16 on the way into LDS; epilogue routes to Q/K/V.
// ---------------------------------------------------------------------------
__global__ __launch_bounds__(256) void qkv_proj_kernel(
    const float* __restrict__ x, const float* __restrict__ wqkv,
    __bf16* __restrict__ Qw, __bf16* __restrict__ Kw, __bf16* __restrict__ Vw) {
  const int f0 = blockIdx.x * 128;
  const int l0 = blockIdx.y * 128;
  const int b  = blockIdx.z;

  __shared__ __bf16 As[128][40];  // [m=f][k=c]
  __shared__ __bf16 Bs[128][40];  // [n=l][k=c]

  const int tid  = threadIdx.x;
  const int lane = tid & 31;
  const int w    = tid >> 5;
  const int wm   = w & 3;   // 4 waves over M: 32 rows each
  const int wn   = w >> 2;  // 2 waves over N: 64 cols each

  v8f acc[2][4] = {};

  for (int k0 = 0; k0 < CIN; k0 += 32) {
#pragma unroll
    for (int i = tid; i < 128 * 32; i += 256) {
      const int cc = i >> 7, rr = i & 127;
      As[rr][cc] = (__bf16)wqkv[(size_t)(k0 + cc) * F3 + f0 + rr];
      Bs[rr][cc] = (__bf16)x[((size_t)b * CIN + k0 + cc) * LSEQ + l0 + rr];
    }
    __syncthreads();

    v16bf a[2];
#pragma unroll
    for (int mi = 0; mi < 2; ++mi)
      a[mi] = load_frag_a(&As[wm * 32 + mi * 16][0], 40, lane);
#pragma unroll
    for (int ni = 0; ni < 4; ++ni) {
      v16bf bf = load_frag_b(&Bs[wn * 64 + ni * 16][0], 40, lane);
#pragma unroll
      for (int mi = 0; mi < 2; ++mi) acc[mi][ni] = wmma_bf16(a[mi], bf, acc[mi][ni]);
    }
    __syncthreads();
  }

  // Epilogue. which = f0>>9 is BLOCK-UNIFORM (128-row tiles never straddle a
  // 512 boundary) -> scalar branch, no per-lane exec juggling.
  const int which = f0 >> 9;  // 0=q 1=k 2=v
  const int fr0   = f0 & 511;
  const int col   = lane & 15;
  const int erow  = (lane & 16) ? 8 : 0;
  const size_t bbase = (size_t)b * HEADS;

  if (which == 0) {
#pragma unroll
    for (int mi = 0; mi < 2; ++mi)
#pragma unroll
      for (int ni = 0; ni < 4; ++ni)
#pragma unroll
        for (int e = 0; e < 8; ++e) {
          const int fr = fr0 + wm * 32 + mi * 16 + erow + e;
          const int l  = l0 + wn * 64 + ni * 16 + col;
          Qw[((bbase + (fr >> 6)) * LSEQ + l) * DHEAD + (fr & 63)] =
              (__bf16)(acc[mi][ni][e] * QK_SCALE);
        }
  } else if (which == 1) {
#pragma unroll
    for (int mi = 0; mi < 2; ++mi)
#pragma unroll
      for (int ni = 0; ni < 4; ++ni)
#pragma unroll
        for (int e = 0; e < 8; ++e) {
          const int fr = fr0 + wm * 32 + mi * 16 + erow + e;
          const int l  = l0 + wn * 64 + ni * 16 + col;
          Kw[((bbase + (fr >> 6)) * LSEQ + l) * DHEAD + (fr & 63)] =
              (__bf16)acc[mi][ni][e];
        }
  } else {
#pragma unroll
    for (int mi = 0; mi < 2; ++mi)
#pragma unroll
      for (int ni = 0; ni < 4; ++ni)
#pragma unroll
        for (int e = 0; e < 8; ++e) {
          const int fr = fr0 + wm * 32 + mi * 16 + erow + e;
          const int l  = l0 + wn * 64 + ni * 16 + col;
          Vw[((bbase + (fr >> 6)) * DHEAD + (fr & 63)) * LSEQ + l] =
              (__bf16)acc[mi][ni][e];
        }
  }
}

// ---------------------------------------------------------------------------
// Stage 2: flash attention per (b,h). Block = 8 waves = 128 query rows.
// K tile (128x64, contiguous 16KB) and V tile (64x128, strided) are staged
// into LDS with async b128 copies. Ks region is ALIASED with the per-wave P
// strips (Ks dead after S-GEMM, P born after it); barriers enforce the order:
//   async fill -> s_wait_asynccnt+bar -> S-GEMM(Ks) -> bar -> softmax/P(Ps)
//   -> O-GEMM(Ps,Vs) -> bar -> next async fill.
// ---------------------------------------------------------------------------
#define KS_LDB 144   // bytes/row: 64 bf16 data + pad (mult of 16)
#define VS_LDB 272   // bytes/row: 128 bf16 data + pad
#define PS_LDB 272   // bytes/row of a P strip
#define PS_WSTRIDE (16 * PS_LDB)          // 4352 B per wave strip
#define SM_PS_BYTES (8 * PS_WSTRIDE)      // 34816 (also covers Ks: 128*144=18432)
#define SM_VS_BYTES (64 * VS_LDB)         // 17408

__global__ __launch_bounds__(256) void flash_attn_kernel(
    const __bf16* __restrict__ Qw, const __bf16* __restrict__ Kw,
    const __bf16* __restrict__ Vw, __bf16* __restrict__ Ow) {
  const int i0 = blockIdx.x * 128;
  const int bh = blockIdx.y;  // b*HEADS + h
  const int b  = bh >> 3, h = bh & 7;

  const int tid  = threadIdx.x;
  const int lane = tid & 31;
  const int w    = tid >> 5;
  const int row0 = i0 + w * 16;

  __shared__ __align__(16) char smem[SM_PS_BYTES + SM_VS_BYTES];
  __bf16* Ps = (__bf16*)smem;                  // [8][16][136] (per-wave strips)
  __bf16* Ks = (__bf16*)smem;                  // [128][72]  (aliases Ps)
  __bf16* Vs = (__bf16*)(smem + SM_PS_BYTES);  // [64][136]
  const unsigned ksBase = lds_addr(Ks);
  const unsigned vsBase = lds_addr(Vs);

  const __bf16* Qb = Qw + (size_t)bh * LSEQ * DHEAD;
  const __bf16* Kb = Kw + (size_t)bh * LSEQ * DHEAD;
  const __bf16* Vb = Vw + (size_t)bh * DHEAD * LSEQ;

  // Q A-fragments for the wave's 16 rows, 2 k-steps over D=64 (held in regs).
  v16bf aq[2];
  {
    const int m  = lane & 15;
    const int kb = (lane & 16) ? 8 : 0;
#pragma unroll
    for (int ks = 0; ks < 2; ++ks) {
      const __bf16* p = Qb + (size_t)(row0 + m) * DHEAD + ks * 32 + kb;
      aq[ks] = frag16(p, p + 16);
    }
  }

  v8f oacc[4] = {};
  float mrow[8], lrow[8];
#pragma unroll
  for (int e = 0; e < 8; ++e) { mrow[e] = -INFINITY; lrow[e] = 0.0f; }

  const int col  = lane & 15;
  const int prow = (lane & 16) ? 8 : 0;

  for (int j0 = 0; j0 < LSEQ; j0 += 128) {
    // ---- async stage K tile (contiguous 16KB) and V tile (64 x 256B) ----
    {
      const char* Kg = (const char*)(Kb + (size_t)j0 * DHEAD);
      const char* Vg = (const char*)Vb + (size_t)j0 * 2;
#pragma unroll
      for (int c = 0; c < 4; ++c) {
        const int idx = tid + c * 256;  // 0..1023 : 16B chunks
        const int jr = idx >> 3, ck = idx & 7;
        async_b128(ksBase + jr * KS_LDB + ck * 16, Kg + (size_t)idx * 16);
        const int dr = idx >> 4, cv = idx & 15;
        async_b128(vsBase + dr * VS_LDB + cv * 16, Vg + (size_t)dr * 4096 + cv * 16);
      }
      wait_async0();
      __syncthreads();
    }

    // ---- S = Q^T K over this 128-col key tile (B-frags from LDS Ks) ----
    v8f sacc[8] = {};
#pragma unroll
    for (int ji = 0; ji < 8; ++ji) {
#pragma unroll
      for (int ks = 0; ks < 2; ++ks) {
        const int kb2 = ((lane & 16) ? 8 : 0) + ks * 32;
        const __bf16* p = Ks + (size_t)(ji * 16 + col) * (KS_LDB / 2) + kb2;
        v16bf bk = frag16(p, p + 16);
        sacc[ji] = wmma_bf16(aq[ks], bk, sacc[ji]);
      }
    }
    __syncthreads();  // all waves done with Ks before P strips overwrite it

    // ---- online softmax (row stats via half-wave shuffles) ----
    float tmax[8];
#pragma unroll
    for (int e = 0; e < 8; ++e) {
      float v = sacc[0][e];
#pragma unroll
      for (int ji = 1; ji < 8; ++ji) v = fmaxf(v, sacc[ji][e]);
      tmax[e] = v;
    }
#pragma unroll
    for (int off = 1; off < 16; off <<= 1)
#pragma unroll
      for (int e = 0; e < 8; ++e) tmax[e] = fmaxf(tmax[e], __shfl_xor(tmax[e], off));

    float alpha[8];
#pragma unroll
    for (int e = 0; e < 8; ++e) {
      const float mn = fmaxf(mrow[e], tmax[e]);
      alpha[e] = __expf(mrow[e] - mn);
      mrow[e]  = mn;
    }

    float tsum[8];
#pragma unroll
    for (int e = 0; e < 8; ++e) tsum[e] = 0.0f;
    __bf16* psw = Ps + w * (PS_WSTRIDE / 2);
#pragma unroll
    for (int ji = 0; ji < 8; ++ji)
#pragma unroll
      for (int e = 0; e < 8; ++e) {
        const float p = __expf(sacc[ji][e] - mrow[e]);
        tsum[e] += p;
        psw[(size_t)(prow + e) * (PS_LDB / 2) + ji * 16 + col] = (__bf16)p;
      }
#pragma unroll
    for (int off = 1; off < 16; off <<= 1)
#pragma unroll
      for (int e = 0; e < 8; ++e) tsum[e] += __shfl_xor(tsum[e], off);
#pragma unroll
    for (int e = 0; e < 8; ++e) lrow[e] = lrow[e] * alpha[e] + tsum[e];
#pragma unroll
    for (int di = 0; di < 4; ++di)
#pragma unroll
      for (int e = 0; e < 8; ++e) oacc[di][e] *= alpha[e];

    // ---- O += P * V^T (A from own P strip, B from LDS Vs) ----
#pragma unroll
    for (int jc = 0; jc < 4; ++jc) {
      const int kb = ((lane & 16) ? 8 : 0) + jc * 32;
      const __bf16* pp = psw + (size_t)col * (PS_LDB / 2) + kb;
      v16bf ap = frag16(pp, pp + 16);
#pragma unroll
      for (int di = 0; di < 4; ++di) {
        const int kb2 = ((lane & 16) ? 8 : 0) + jc * 32;
        const __bf16* pv = Vs + (size_t)(di * 16 + col) * (VS_LDB / 2) + kb2;
        v16bf bv = frag16(pv, pv + 16);
        oacc[di] = wmma_bf16(ap, bv, oacc[di]);
      }
    }
    __syncthreads();  // Ps/Vs fully consumed before next tile's async fill
  }

  // Finalize: O[b][l][h*64+d] = oacc / l
#pragma unroll
  for (int di = 0; di < 4; ++di)
#pragma unroll
    for (int e = 0; e < 8; ++e) {
      const int irow = row0 + prow + e;
      const int d = di * 16 + col;
      const float val = oacc[di][e] / lrow[e];
      Ow[((size_t)b * LSEQ + irow) * HD + h * DHEAD + d] = (__bf16)val;
    }
}

// ---------------------------------------------------------------------------
// Stage 3: out[b][f][l] = sum_c O[b][l][c] * wout[c][f] + bout[f]
// (M=f 512, N=l 2048, K=c 512), fp32 output. B tile (bf16) staged via async.
// ---------------------------------------------------------------------------
__global__ __launch_bounds__(256) void out_proj_kernel(
    const __bf16* __restrict__ Ow, const float* __restrict__ wout,
    const float* __restrict__ bout, float* __restrict__ out) {
  const int f0 = blockIdx.x * 128;
  const int l0 = blockIdx.y * 128;
  const int b  = blockIdx.z;

  __shared__ __bf16 As[128][40];                  // [m=f][k=c] (f32->bf16 convert)
  __shared__ __align__(16) __bf16 Bs[128][40];    // [n=l][k=c] (async-staged)
  const unsigned bsBase = lds_addr(&Bs[0][0]);

  const int tid  = threadIdx.x;
  const int lane = tid & 31;
  const int w    = tid >> 5;
  const int wm   = w & 3;
  const int wn   = w >> 2;

  v8f acc[2][4] = {};

  for (int k0 = 0; k0 < HD; k0 += 32) {
    // B tile: 128 rows x 64B via async b128 (2 chunks per thread)
    {
      const char* Og = (const char*)Ow + (((size_t)b * LSEQ + l0) * HD + k0) * 2;
#pragma unroll
      for (int c = 0; c < 2; ++c) {
        const int idx = tid + c * 256;  // 0..511
        const int ll = idx >> 2, ck = idx & 3;
        async_b128(bsBase + ll * 80 + ck * 16, Og + (size_t)ll * (HD * 2) + ck * 16);
      }
    }
    // A tile: f32 -> bf16 conversion in-register
#pragma unroll
    for (int i = tid; i < 128 * 32; i += 256) {
      const int cc = i >> 7, rr = i & 127;
      As[rr][cc] = (__bf16)wout[(size_t)(k0 + cc) * HD + f0 + rr];
    }
    wait_async0();
    __syncthreads();

    v16bf a[2];
#pragma unroll
    for (int mi = 0; mi < 2; ++mi)
      a[mi] = load_frag_a(&As[wm * 32 + mi * 16][0], 40, lane);
#pragma unroll
    for (int ni = 0; ni < 4; ++ni) {
      v16bf bf = load_frag_b(&Bs[wn * 64 + ni * 16][0], 40, lane);
#pragma unroll
      for (int mi = 0; mi < 2; ++mi) acc[mi][ni] = wmma_bf16(a[mi], bf, acc[mi][ni]);
    }
    __syncthreads();
  }

  const int col  = lane & 15;
  const int erow = (lane & 16) ? 8 : 0;
#pragma unroll
  for (int mi = 0; mi < 2; ++mi)
#pragma unroll
    for (int ni = 0; ni < 4; ++ni)
#pragma unroll
      for (int e = 0; e < 8; ++e) {
        const int f = f0 + wm * 32 + mi * 16 + erow + e;
        const int l = l0 + wn * 64 + ni * 16 + col;
        out[((size_t)b * CIN + f) * LSEQ + l] = acc[mi][ni][e] + bout[f];
      }
}

// ---------------------------------------------------------------------------
extern "C" void kernel_launch(void* const* d_in, const int* in_sizes, int n_in,
                              void* d_out, int out_size, void* d_ws, size_t ws_size,
                              hipStream_t stream) {
  const float* x    = (const float*)d_in[0];
  const float* wqkv = (const float*)d_in[1];
  const float* wout = (const float*)d_in[2];
  const float* bout = (const float*)d_in[3];
  float* out = (float*)d_out;

  char* ws = (char*)d_ws;
  const size_t SEG = (size_t)BATCH * HEADS * LSEQ * DHEAD * sizeof(__bf16);  // 8MB
  __bf16* Qw = (__bf16*)(ws);
  __bf16* Kw = (__bf16*)(ws + SEG);
  __bf16* Vw = (__bf16*)(ws + 2 * SEG);
  __bf16* Ow = (__bf16*)(ws + 3 * SEG);

  qkv_proj_kernel<<<dim3(F3 / 128, LSEQ / 128, BATCH), 256, 0, stream>>>(
      x, wqkv, Qw, Kw, Vw);
  flash_attn_kernel<<<dim3(LSEQ / 128, BATCH * HEADS), 256, 0, stream>>>(
      Qw, Kw, Vw, Ow);
  out_proj_kernel<<<dim3(HD / 128, LSEQ / 128, BATCH), 256, 0, stream>>>(
      Ow, wout, bout, out);
}
